// GCNNet_8108898255427
// MI455X (gfx1250) — matrix operand order repaired and checked by reading, not compiled, not run on hardware
//
#include <hip/hip_runtime.h>
#include <hip/hip_bf16.h>
#include <math.h>

typedef _Float16 f16;
typedef f16   v16h __attribute__((ext_vector_type(16)));
typedef f16   v8h  __attribute__((ext_vector_type(8)));
typedef float v8f  __attribute__((ext_vector_type(8)));

#define HH     2
#define EE     3
#define NN     3072
#define D_IN   256
#define D_H    128
#define D_OUTD 64
#define D_PRED 64
#define CC     8
#define MM     512
#define EPSV   1e-5f
#define NEGV   -9e15f

// ---------------- WMMA fragment loaders (16x16x32 f16, wave32) ----------------
// A 16x32 (MxK): lane<16 -> row=lane,   elems 0..7 = K 0..7,  8..15 = K 16..23
//                lane>=16 -> row=lane-16, elems 0..7 = K 8..15, 8..15 = K 24..31
__device__ __forceinline__ v16h load_a_frag(const f16* A, int lda, int row0, int k0, int lane) {
    int r  = row0 + (lane & 15);
    int kh = (lane >> 4) * 8;
    const f16* p = A + (size_t)r * lda + k0 + kh;
    v8h lo = *(const v8h*)p;
    v8h hi = *(const v8h*)(p + 16);
    v16h a;
#pragma unroll
    for (int i = 0; i < 8; i++) { a[i] = lo[i]; a[8 + i] = hi[i]; }
    return a;
}

// B 32x16 (KxN) for A @ Bm^T where Bm is row-major [ncols, K] (i.e. K contiguous):
// lane<16: col=lane, elems 0..7 = K 0..7, 8..15 = K 16..23 ; lane>=16: col=lane-16, elems 0..7 = K 8..15, 8..15 = K 24..31
__device__ __forceinline__ v16h load_bt_frag(const f16* Bm, int ldb, int col0, int k0, int lane) {
    int c  = col0 + (lane & 15);
    int kh = (lane >> 4) * 8;
    const f16* p = Bm + (size_t)c * ldb + k0 + kh;
    v8h lo = *(const v8h*)p;
    v8h hi = *(const v8h*)(p + 16);
    v16h b;
#pragma unroll
    for (int i = 0; i < 8; i++) { b[i] = lo[i]; b[8 + i] = hi[i]; }
    return b;
}

__device__ __forceinline__ v8f wmma_f16(v16h a, v16h b, v8f c) {
    return __builtin_amdgcn_wmma_f32_16x16x32_f16(false, a, false, b, (short)0, c, false, false);
}

__device__ __forceinline__ float eluf(float v) { return v > 0.f ? v : __expf(v) - 1.f; }

// ---------------- kernels ----------------
__global__ void k_cvt(const float* __restrict__ s, f16* __restrict__ d, int n) {
    int i = blockIdx.x * blockDim.x + threadIdx.x;
    if (i < n) d[i] = (f16)s[i];
}

// src [H, R, C] f32 -> dst [H, C, R] f16   (weight transpose so B-fragments are contiguous)
__global__ void k_cvtT(const float* __restrict__ src, f16* __restrict__ dst, int R, int C) {
    int i = blockIdx.x * blockDim.x + threadIdx.x;
    int total = HH * R * C;
    if (i >= total) return;
    int h = i / (R * C);
    int rem = i - h * (R * C);
    int r = rem / C, c = rem - r * C;
    dst[(size_t)h * R * C + (size_t)c * R + r] = (f16)src[i];
}

// Y[h,n,dout] = X[(h),n,:] @ Wt[h,dout,:]^T + bias[h,dout]   (one wave per 16x16 tile)
__global__ void k_lin(const f16* __restrict__ X, const f16* __restrict__ Wt,
                      const float* __restrict__ bias, float* __restrict__ Y,
                      int K, int Dout, int xPerHead) {
    int lane = threadIdx.x;
    int n0 = blockIdx.x * 16, c0 = blockIdx.y * 16, h = blockIdx.z;
    const f16* Xh  = X + (xPerHead ? (size_t)h * NN * K : 0);
    const f16* Wth = Wt + (size_t)h * K * Dout;   // [Dout, K]
    v8f acc = {0.f, 0.f, 0.f, 0.f, 0.f, 0.f, 0.f, 0.f};
    for (int k0 = 0; k0 < K; k0 += 32) {
        v16h a = load_a_frag(Xh, K, n0, k0, lane);
        v16h b = load_bt_frag(Wth, K, c0, k0, lane);
        acc = wmma_f16(a, b, acc);
    }
    int c = c0 + (lane & 15);
    int rb = n0 + (lane >> 4) * 8;
    float bi = bias[h * Dout + c];
    float* Yh = Y + (size_t)h * NN * Dout;
#pragma unroll
    for (int i = 0; i < 8; i++) Yh[(size_t)(rb + i) * Dout + c] = acc[i] + bi;
}

// BatchNorm over node axis for one (h,d) column; writes f16 h (row-major) and h^T
__global__ void k_bn(const float* __restrict__ X, const float* __restrict__ g,
                     const float* __restrict__ b, f16* __restrict__ Yh,
                     f16* __restrict__ YhT, int D) {
    int col = blockIdx.x;            // h*D + d
    int h = col / D, d = col % D;
    const float* Xc = X + (size_t)h * NN * D + d;
    __shared__ float red[256];
    int tid = threadIdx.x;
    float s = 0.f, sq = 0.f;
    for (int n = tid; n < NN; n += 256) {
        float v = Xc[(size_t)n * D];
        s += v; sq += v * v;
    }
    red[tid] = s; __syncthreads();
    for (int o = 128; o > 0; o >>= 1) { if (tid < o) red[tid] += red[tid + o]; __syncthreads(); }
    float sum = red[0]; __syncthreads();
    red[tid] = sq; __syncthreads();
    for (int o = 128; o > 0; o >>= 1) { if (tid < o) red[tid] += red[tid + o]; __syncthreads(); }
    float sumsq = red[0]; __syncthreads();
    float mu  = sum / NN;
    float var = sumsq / NN - mu * mu;   // biased, matches jnp.var
    float gg  = g[col] * rsqrtf(var + EPSV);
    float bb  = b[col];
    f16* Yc  = Yh + (size_t)h * NN * D + d;
    f16* Ytc = YhT + ((size_t)h * D + d) * NN;   // contiguous along n
    for (int n = tid; n < NN; n += 256) {
        f16 v = (f16)((Xc[(size_t)n * D] - mu) * gg + bb);
        Yc[(size_t)n * D] = v;
        Ytc[n] = v;
    }
}

// Fused: scores (WMMA) -> LeakyReLU -> mask -> per-edge softmax -> edge-weighted
// accumulate into alpha[h,n,:].  One block = one (h, 16-row) tile, 8 waves.
template <int DK>
__global__ void k_scores(const f16* __restrict__ Hbn, const f16* __restrict__ AttW,
                         const float* __restrict__ attB, const float* __restrict__ headW,
                         const unsigned char* __restrict__ mask, float* __restrict__ alpha) {
    const int D = DK * 32;
    extern __shared__ float srow[];   // [16][NN]
    int tid = threadIdx.x, lane = tid & 31, wave = tid >> 5;
    int n0 = blockIdx.x * 16, h = blockIdx.y;

    float w0 = headW[h * EE + 0], w1 = headW[h * EE + 1], w2 = headW[h * EE + 2];
    float wm = fmaxf(w0, fmaxf(w1, w2));
    float ew[EE] = {__expf(w0 - wm), __expf(w1 - wm), __expf(w2 - wm)};
    float ews = ew[0] + ew[1] + ew[2];

    v16h afr[DK];
#pragma unroll
    for (int kk = 0; kk < DK; kk++)
        afr[kk] = load_a_frag(Hbn + (size_t)h * NN * D, D, n0, kk * 32, lane);

    for (int e = 0; e < EE; e++) {
        const f16* Bm = AttW + (size_t)(h * EE + e) * NN * D;   // [N, D], K contiguous
        const float* bias = attB + (size_t)(h * EE + e) * NN;
        for (int mt = wave; mt < NN / 16; mt += 8) {
            int m0 = mt * 16;
            v8f acc = {0.f, 0.f, 0.f, 0.f, 0.f, 0.f, 0.f, 0.f};
#pragma unroll
            for (int kk = 0; kk < DK; kk++) {
                v16h b = load_bt_frag(Bm, D, m0, kk * 32, lane);
                acc = wmma_f16(afr[kk], b, acc);
            }
            int c = lane & 15, rb = (lane >> 4) * 8;
            float bc = bias[m0 + c];
#pragma unroll
            for (int i = 0; i < 8; i++) {
                int r = rb + i;
                float v = acc[i] + bc;
                v = v > 0.f ? v : 0.2f * v;                       // LeakyReLU(0.2)
                bool mk = mask[((size_t)e * NN + (n0 + r)) * NN + m0 + c] != 0;
                srow[r * NN + m0 + c] = mk ? v : -1e30f;          // exp underflow -> exact 0
            }
        }
        __syncthreads();
        float we = ew[e] / ews;
        for (int r = wave; r < 16; r += 8) {
            const float* sr = srow + r * NN;
            float mx = -3.4e38f;
            for (int j = lane; j < NN; j += 32) mx = fmaxf(mx, sr[j]);
#pragma unroll
            for (int o = 16; o > 0; o >>= 1) mx = fmaxf(mx, __shfl_xor(mx, o, 32));
            float sum = 0.f;
            for (int j = lane; j < NN; j += 32) sum += __expf(sr[j] - mx);
#pragma unroll
            for (int o = 16; o > 0; o >>= 1) sum += __shfl_xor(sum, o, 32);
            float scale = we / sum;
            float* ar = alpha + ((size_t)h * NN + n0 + r) * NN;
            for (int j = lane; j < NN; j += 32) {
                float val = __expf(sr[j] - mx) * scale;
                ar[j] = (e == 0) ? val : (ar[j] + val);
            }
        }
        __syncthreads();
    }
}

// alpha2 = softmax(where(alpha>0, alpha, NEG)) per row; emit f16
__global__ void k_soft2(const float* __restrict__ alpha, f16* __restrict__ alphaH) {
    size_t row = blockIdx.x;
    const float* a = alpha + row * NN;
    f16* oh = alphaH + row * NN;
    __shared__ float red[256];
    int tid = threadIdx.x;
    float mx = -3.4e38f;
    for (int j = tid; j < NN; j += 256) { float v = a[j]; v = v > 0.f ? v : NEGV; mx = fmaxf(mx, v); }
    red[tid] = mx; __syncthreads();
    for (int o = 128; o > 0; o >>= 1) { if (tid < o) red[tid] = fmaxf(red[tid], red[tid + o]); __syncthreads(); }
    mx = red[0]; __syncthreads();
    float s = 0.f;
    for (int j = tid; j < NN; j += 256) { float v = a[j]; v = v > 0.f ? v : NEGV; s += __expf(v - mx); }
    red[tid] = s; __syncthreads();
    for (int o = 128; o > 0; o >>= 1) { if (tid < o) red[tid] += red[tid + o]; __syncthreads(); }
    float inv = 1.f / red[0];
    for (int j = tid; j < NN; j += 256) {
        float v = a[j]; v = v > 0.f ? v : NEGV;
        oh[j] = (f16)(__expf(v - mx) * inv);
    }
}

// Out[h,n,d] = alpha[h,n,:] @ Hbn[h,:,d]  using transposed HbnT [H,D,N].
// One wave computes a 16x32 output tile (two col tiles reuse one A fragment).
__global__ void k_av(const f16* __restrict__ alphaH, const f16* __restrict__ HbnT,
                     float* __restrict__ Out, f16* __restrict__ Out16, int D, int doElu) {
    int lane = threadIdx.x;
    int n0 = blockIdx.x * 16, c0 = blockIdx.y * 32, h = blockIdx.z;
    const f16* A  = alphaH + (size_t)h * NN * NN;
    const f16* Bt = HbnT + (size_t)h * D * NN;    // [D, N], K contiguous
    v8f acc0 = {0.f, 0.f, 0.f, 0.f, 0.f, 0.f, 0.f, 0.f};
    v8f acc1 = acc0;
    for (int k0 = 0; k0 < NN; k0 += 32) {
        if (k0 + 64 < NN) {
            __builtin_prefetch(A + (size_t)(n0 + (lane & 15)) * NN + k0 + 64, 0, 0);
            __builtin_prefetch(Bt + (size_t)(c0 + (lane & 15)) * NN + k0 + 64, 0, 0);
        }
        v16h a  = load_a_frag(A, NN, n0, k0, lane);
        v16h b0 = load_bt_frag(Bt, NN, c0, k0, lane);
        v16h b1 = load_bt_frag(Bt, NN, c0 + 16, k0, lane);
        acc0 = wmma_f16(a, b0, acc0);
        acc1 = wmma_f16(a, b1, acc1);
    }
    int rb = n0 + (lane >> 4) * 8;
#pragma unroll
    for (int t = 0; t < 2; t++) {
        v8f acc = t ? acc1 : acc0;
        int c = c0 + t * 16 + (lane & 15);
#pragma unroll
        for (int i = 0; i < 8; i++) {
            float v = acc[i];
            if (doElu) v = eluf(v);
            size_t idx = (size_t)h * NN * D + (size_t)(rb + i) * D + c;
            Out[idx] = v;
            if (Out16) Out16[idx] = (f16)v;
        }
    }
}

// comb = (head0 + head1)/2 ; stats[d] = mean, stats[64+d] = rsqrt(var+eps)
__global__ void k_finalbn(const float* __restrict__ Out1, float* __restrict__ comb,
                          float* __restrict__ stats) {
    int d = blockIdx.x;
    __shared__ float red[256];
    int tid = threadIdx.x;
    float s = 0.f, sq = 0.f;
    for (int n = tid; n < NN; n += 256) {
        float v = (Out1[(size_t)n * D_OUTD + d] + Out1[(size_t)(NN + n) * D_OUTD + d]) * 0.5f;
        comb[(size_t)n * D_OUTD + d] = v;
        s += v; sq += v * v;
    }
    red[tid] = s; __syncthreads();
    for (int o = 128; o > 0; o >>= 1) { if (tid < o) red[tid] += red[tid + o]; __syncthreads(); }
    float sum = red[0]; __syncthreads();
    red[tid] = sq; __syncthreads();
    for (int o = 128; o > 0; o >>= 1) { if (tid < o) red[tid] += red[tid + o]; __syncthreads(); }
    if (tid == 0) {
        float mu = sum / NN;
        float var = red[0] / NN - mu * mu;
        stats[d] = mu;
        stats[D_OUTD + d] = rsqrtf(var + EPSV);
    }
}

// gather target rows -> BN -> ELU -> MLP(64->64 ELU ->8) -> logits + mean NLL
__global__ void k_pred(const float* __restrict__ comb, const float* __restrict__ stats,
                       const float* __restrict__ g, const float* __restrict__ b,
                       const float* __restrict__ p1W, const float* __restrict__ p1b,
                       const float* __restrict__ p2W, const float* __restrict__ p2b,
                       const int* __restrict__ tX, const int* __restrict__ tgt,
                       float* __restrict__ out) {
    __shared__ float red[256];
    int tid = threadIdx.x;
    float lossAcc = 0.f;
    for (int i = tid; i < MM; i += 256) {
        int node = tX[i];
        float feat[D_OUTD];
#pragma unroll 8
        for (int d = 0; d < D_OUTD; d++) {
            float v = (comb[(size_t)node * D_OUTD + d] - stats[d]) * stats[D_OUTD + d] * g[d] + b[d];
            feat[d] = eluf(v);
        }
        float lg[CC];
#pragma unroll
        for (int c = 0; c < CC; c++) lg[c] = p2b[c];
        for (int o = 0; o < D_PRED; o++) {
            float hv = p1b[o];
#pragma unroll 8
            for (int d = 0; d < D_OUTD; d++) hv += feat[d] * p1W[d * D_PRED + o];
            hv = eluf(hv);
#pragma unroll
            for (int c = 0; c < CC; c++) lg[c] += hv * p2W[o * CC + c];
        }
        float mx = lg[0];
#pragma unroll
        for (int c = 1; c < CC; c++) mx = fmaxf(mx, lg[c]);
        float se = 0.f;
#pragma unroll
        for (int c = 0; c < CC; c++) se += __expf(lg[c] - mx);
        float lse = mx + logf(se);
#pragma unroll
        for (int c = 0; c < CC; c++) out[1 + i * CC + c] = lg[c];
        lossAcc += (lse - lg[tgt[i]]);
    }
    red[tid] = lossAcc; __syncthreads();
    for (int o = 128; o > 0; o >>= 1) { if (tid < o) red[tid] += red[tid + o]; __syncthreads(); }
    if (tid == 0) out[0] = red[0] / (float)MM;
}

// ---------------- host ----------------
extern "C" void kernel_launch(void* const* d_in, const int* in_sizes, int n_in,
                              void* d_out, int out_size, void* d_ws, size_t ws_size,
                              hipStream_t stream) {
    (void)in_sizes; (void)n_in; (void)out_size; (void)ws_size;
    const float* x   = (const float*)d_in[0];
    const float* fW0 = (const float*)d_in[1];
    const float* fb0 = (const float*)d_in[2];
    const float* g0  = (const float*)d_in[3];
    const float* b0  = (const float*)d_in[4];
    const float* aW0 = (const float*)d_in[5];
    const float* ab0 = (const float*)d_in[6];
    const float* hw0 = (const float*)d_in[7];
    const float* fW1 = (const float*)d_in[8];
    const float* fb1 = (const float*)d_in[9];
    const float* g1  = (const float*)d_in[10];
    const float* b1  = (const float*)d_in[11];
    const float* aW1 = (const float*)d_in[12];
    const float* ab1 = (const float*)d_in[13];
    const float* hw1 = (const float*)d_in[14];
    const float* bng = (const float*)d_in[15];
    const float* bnb = (const float*)d_in[16];
    const float* p1W = (const float*)d_in[17];
    const float* p1b = (const float*)d_in[18];
    const float* p2W = (const float*)d_in[19];
    const float* p2b = (const float*)d_in[20];
    const unsigned char* mask = (const unsigned char*)d_in[21];  // numpy bool
    const int* tX  = (const int*)d_in[22];
    const int* tgt = (const int*)d_in[23];
    float* out = (float*)d_out;

    size_t off = 0;
    auto alloc = [&](size_t bytes) -> void* {
        void* p = (char*)d_ws + off;
        off = (off + bytes + 255) & ~(size_t)255;
        return p;
    };
    f16*   x16    = (f16*)alloc((size_t)NN * D_IN * 2);
    f16*   W0t    = (f16*)alloc((size_t)HH * D_IN * D_H * 2);      // [H, D_H, D_IN]
    f16*   W1t    = (f16*)alloc((size_t)HH * D_H * D_OUTD * 2);    // [H, D_OUTD, D_H]
    f16*   AW016  = (f16*)alloc((size_t)HH * EE * NN * D_H * 2);
    f16*   AW116  = (f16*)alloc((size_t)HH * EE * NN * D_OUTD * 2);
    float* hlin   = (float*)alloc((size_t)HH * NN * D_H * 4);
    f16*   hbn16  = (f16*)alloc((size_t)HH * NN * D_H * 2);
    f16*   hbnT   = (f16*)alloc((size_t)HH * NN * D_H * 2);        // [H, D, N]
    float* alpha  = (float*)alloc((size_t)HH * NN * NN * 4);
    f16*   alphaH = (f16*)alloc((size_t)HH * NN * NN * 2);
    f16*   out016 = (f16*)alloc((size_t)HH * NN * D_H * 2);
    float* out1   = (float*)alloc((size_t)HH * NN * D_OUTD * 4);
    float* comb   = (float*)alloc((size_t)NN * D_OUTD * 4);
    float* stats  = (float*)alloc((size_t)2 * D_OUTD * 4);

    auto cvt = [&](const float* s, f16* d, int n) {
        k_cvt<<<(n + 255) / 256, 256, 0, stream>>>(s, d, n);
    };
    cvt(x, x16, NN * D_IN);
    cvt(aW0, AW016, HH * EE * NN * D_H);
    cvt(aW1, AW116, HH * EE * NN * D_OUTD);
    k_cvtT<<<(HH * D_IN * D_H + 255) / 256, 256, 0, stream>>>(fW0, W0t, D_IN, D_H);
    k_cvtT<<<(HH * D_H * D_OUTD + 255) / 256, 256, 0, stream>>>(fW1, W1t, D_H, D_OUTD);

    const size_t smem = (size_t)16 * NN * 4;  // 192KB row buffer (<=320KB/WGP)

    // ---- layer 0 ----
    k_lin<<<dim3(NN / 16, D_H / 16, HH), 32, 0, stream>>>(x16, W0t, fb0, hlin, D_IN, D_H, 0);
    k_bn<<<HH * D_H, 256, 0, stream>>>(hlin, g0, b0, hbn16, hbnT, D_H);
    k_scores<4><<<dim3(NN / 16, HH), 256, smem, stream>>>(hbn16, AW016, ab0, hw0, mask, alpha);
    k_soft2<<<HH * NN, 256, 0, stream>>>(alpha, alphaH);
    k_av<<<dim3(NN / 16, D_H / 32, HH), 32, 0, stream>>>(alphaH, hbnT, hlin, out016, D_H, 1);

    // ---- layer 1 ----
    k_lin<<<dim3(NN / 16, D_OUTD / 16, HH), 32, 0, stream>>>(out016, W1t, fb1, hlin, D_H, D_OUTD, 1);
    k_bn<<<HH * D_OUTD, 256, 0, stream>>>(hlin, g1, b1, hbn16, hbnT, D_OUTD);
    k_scores<2><<<dim3(NN / 16, HH), 256, smem, stream>>>(hbn16, AW116, ab1, hw1, mask, alpha);
    k_soft2<<<HH * NN, 256, 0, stream>>>(alpha, alphaH);
    k_av<<<dim3(NN / 16, D_OUTD / 32, HH), 32, 0, stream>>>(alphaH, hbnT, out1, nullptr, D_OUTD, 0);

    // ---- head combine + final BN + prediction/loss ----
    k_finalbn<<<D_OUTD, 256, 0, stream>>>(out1, comb, stats);
    k_pred<<<1, 256, 0, stream>>>(comb, stats, bng, bnb, p1W, p1b, p2W, p2b, tX, tgt, out);
}
